// ARWKV_7_23880018166088
// MI455X (gfx1250) — matrix operand order repaired and checked by reading, not compiled, not run on hardware
//
#include <hip/hip_runtime.h>
#include <hip/hip_bf16.h>

typedef __bf16 bf16;
typedef __attribute__((ext_vector_type(16))) __bf16 v16bf;
typedef __attribute__((ext_vector_type(8)))  float  v8f;

#define LDT 72   // LDS row stride (bf16 elems) for BK=64: 144B rows, 16B aligned, conflict-skewed

// ---------------------------------------------------------------------------
// bf16 WMMA GEMM:  C[M,N] = A[M,K] @ W[K,N], fp32 out, row-major.
// Block tile 128x128, K-step 64, 256 threads = 8 waves, wave tile 32x64.
// Double-buffered LDS ping-pong: prefetch tile i+1 into regs while WMMAs
// consume tile i; single barrier per K-step. M%128==0, K%32==0 assumed
// (zero-padded to 64); N guarded.
// ---------------------------------------------------------------------------
__global__ __launch_bounds__(256)
void gemm_bf16_kernel(const bf16* __restrict__ A, const bf16* __restrict__ W,
                      float* __restrict__ C, int M, int N, int K) {
    __shared__ bf16 As[2][128 * LDT];
    __shared__ bf16 Bt[2][128 * LDT];   // transposed: Bt[n][k]

    const int tid  = threadIdx.x;
    const int lane = tid & 31;
    const int wave = tid >> 5;
    const int bm   = blockIdx.x * 128;
    const int bn   = blockIdx.y * 128;
    const int wm   = (wave & 3) * 32;
    const int wn   = (wave >> 2) * 64;

    v8f acc[2][4];
#pragma unroll
    for (int i = 0; i < 2; ++i)
#pragma unroll
        for (int j = 0; j < 4; ++j) acc[i][j] = (v8f)0.f;

    const int arow = tid >> 1;           // 0..127
    const int acol = (tid & 1) * 32;     // 0 or 32
    const int brow = tid >> 2;           // 0..63  (k within tile)
    const int bcol = (tid & 3) * 32;     // 0..96  (n within tile)

    const int nt = (K + 63) / 64;

    // ---- prologue: stage tile 0 into buffer 0 ----
    {
        uint4 pa[4];
        const int gka = acol;            // K%32==0 -> 32-chunk fully in/out
        if (gka < K) {
            const bf16* src = A + (size_t)(bm + arow) * K + gka;
            pa[0] = ((const uint4*)src)[0]; pa[1] = ((const uint4*)src)[1];
            pa[2] = ((const uint4*)src)[2]; pa[3] = ((const uint4*)src)[3];
        } else {
            pa[0] = pa[1] = pa[2] = pa[3] = make_uint4(0, 0, 0, 0);
        }
        *(uint4*)(&As[0][arow * LDT + acol])      = pa[0];
        *(uint4*)(&As[0][arow * LDT + acol + 8])  = pa[1];
        *(uint4*)(&As[0][arow * LDT + acol + 16]) = pa[2];
        *(uint4*)(&As[0][arow * LDT + acol + 24]) = pa[3];

        bf16 pb[32];
        const int  gk  = brow;
        const bool kin = gk < K;
        const bf16* src = W + (size_t)gk * N + bn + bcol;
        if (kin && bn + bcol + 32 <= N) {
            *(uint4*)(pb + 0)  = ((const uint4*)src)[0];
            *(uint4*)(pb + 8)  = ((const uint4*)src)[1];
            *(uint4*)(pb + 16) = ((const uint4*)src)[2];
            *(uint4*)(pb + 24) = ((const uint4*)src)[3];
        } else {
#pragma unroll
            for (int j = 0; j < 32; ++j)
                pb[j] = (kin && bn + bcol + j < N) ? src[j] : (bf16)0.f;
        }
#pragma unroll
        for (int j = 0; j < 32; ++j) Bt[0][(bcol + j) * LDT + brow] = pb[j];
    }
    __syncthreads();

    int cur = 0;
    for (int it = 0; it < nt; ++it) {
        // ---- prefetch next tile into registers (latency hidden by WMMAs) ----
        const bool hasnext = (it + 1) < nt;
        uint4 pa[4];
        bf16  pb[32];
        if (hasnext) {
            const int k1  = (it + 1) * 64;
            const int gka = k1 + acol;
            if (gka < K) {
                const bf16* src = A + (size_t)(bm + arow) * K + gka;
                pa[0] = ((const uint4*)src)[0]; pa[1] = ((const uint4*)src)[1];
                pa[2] = ((const uint4*)src)[2]; pa[3] = ((const uint4*)src)[3];
            } else {
                pa[0] = pa[1] = pa[2] = pa[3] = make_uint4(0, 0, 0, 0);
            }
            const int  gk  = k1 + brow;
            const bool kin = gk < K;
            const bf16* src = W + (size_t)gk * N + bn + bcol;
            if (kin && bn + bcol + 32 <= N) {
                *(uint4*)(pb + 0)  = ((const uint4*)src)[0];
                *(uint4*)(pb + 8)  = ((const uint4*)src)[1];
                *(uint4*)(pb + 16) = ((const uint4*)src)[2];
                *(uint4*)(pb + 24) = ((const uint4*)src)[3];
            } else {
#pragma unroll
                for (int j = 0; j < 32; ++j)
                    pb[j] = (kin && bn + bcol + j < N) ? src[j] : (bf16)0.f;
            }
        }

        // ---- compute from current buffer: 2 sub-steps of K=32, 16 WMMAs ----
        const int mrow = lane & 15;
        const int kbA  = (lane < 16) ? 0 : 8;    // A frag: lanes 0-15 hold K 0-7/16-23
        const int kbB  = (lane < 16) ? 0 : 8;    // B frag: lanes 0-15 hold K 0-7/16-23
#pragma unroll
        for (int kk = 0; kk < 64; kk += 32) {
            v16bf afrag[2], bfrag[4];
#pragma unroll
            for (int mi = 0; mi < 2; ++mi) {
                union { v16bf v; unsigned u[8]; } ua;
                const unsigned* p =
                    (const unsigned*)&As[cur][(wm + mi * 16 + mrow) * LDT + kk];
#pragma unroll
                for (int j = 0; j < 4; ++j) {
                    ua.u[j]     = p[(kbA >> 1) + j];
                    ua.u[4 + j] = p[((kbA + 16) >> 1) + j];
                }
                afrag[mi] = ua.v;
            }
#pragma unroll
            for (int ni = 0; ni < 4; ++ni) {
                union { v16bf v; unsigned u[8]; } ub;
                const unsigned* p =
                    (const unsigned*)&Bt[cur][(wn + ni * 16 + (lane & 15)) * LDT + kk];
#pragma unroll
                for (int j = 0; j < 4; ++j) {
                    ub.u[j]     = p[(kbB >> 1) + j];
                    ub.u[4 + j] = p[((kbB + 16) >> 1) + j];
                }
                bfrag[ni] = ub.v;
            }
#pragma unroll
            for (int mi = 0; mi < 2; ++mi)
#pragma unroll
                for (int ni = 0; ni < 4; ++ni)
                    acc[mi][ni] = __builtin_amdgcn_wmma_f32_16x16x32_bf16(
                        false, afrag[mi], false, bfrag[ni], (short)0, acc[mi][ni],
                        false, false);
        }

        // ---- commit prefetched tile to the other buffer ----
        if (hasnext) {
            const int nxt = cur ^ 1;
            *(uint4*)(&As[nxt][arow * LDT + acol])      = pa[0];
            *(uint4*)(&As[nxt][arow * LDT + acol + 8])  = pa[1];
            *(uint4*)(&As[nxt][arow * LDT + acol + 16]) = pa[2];
            *(uint4*)(&As[nxt][arow * LDT + acol + 24]) = pa[3];
#pragma unroll
            for (int j = 0; j < 32; ++j) Bt[nxt][(bcol + j) * LDT + brow] = pb[j];
        }
        __syncthreads();
        cur ^= 1;
    }

    // ---- writeback: C layout: lane = N col, VGPR p = row p (+8 for hi lanes)
    const int col0 = lane & 15;
    const int rowh = (lane < 16) ? 0 : 8;
#pragma unroll
    for (int mi = 0; mi < 2; ++mi)
#pragma unroll
        for (int ni = 0; ni < 4; ++ni) {
            int gcol = bn + wn + ni * 16 + col0;
            if (gcol >= N) continue;
#pragma unroll
            for (int p = 0; p < 8; ++p) {
                int grow = bm + wm + mi * 16 + rowh + p;
                C[(size_t)grow * N + gcol] = acc[mi][ni][p];
            }
        }
}

// ---------------------------------------------------------------------------
__global__ void cvt_f32_bf16(const float* __restrict__ in, bf16* __restrict__ out, int n) {
    int i = blockIdx.x * blockDim.x + threadIdx.x;
    if (i < n) out[i] = (bf16)in[i];
}

// fp32 -> bf16 with optional activation: 0=id, 1=tanh, 2=sigmoid
__global__ void act_cvt(const float* __restrict__ in, bf16* __restrict__ out, int n, int mode) {
    int i = blockIdx.x * blockDim.x + threadIdx.x;
    if (i >= n) return;
    float v = in[i];
    if (mode == 1) v = tanhf(v);
    else if (mode == 2) v = 1.f / (1.f + expf(-v));
    out[i] = (bf16)v;
}

// ---------------------------------------------------------------------------
// rmsnorm(ln1) + token shift -> six bf16 shifted activations; also xn last row
// ---------------------------------------------------------------------------
__global__ __launch_bounds__(256)
void rmsnorm_shift(const float* __restrict__ x, const float* __restrict__ x_prev,
                   const float* __restrict__ ln1w,
                   const float* __restrict__ mr, const float* __restrict__ mw,
                   const float* __restrict__ mk, const float* __restrict__ mv,
                   const float* __restrict__ ma, const float* __restrict__ mg,
                   bf16* xr, bf16* xw, bf16* xk, bf16* xv, bf16* xa, bf16* xg,
                   float* __restrict__ xn_last, int T, int D) {
    const int t = blockIdx.x, tid = threadIdx.x;
    __shared__ float sbuf[256];

    float s = 0.f, sp = 0.f;
    for (int d = tid; d < D; d += 256) {
        float v = x[(size_t)t * D + d]; s += v * v;
        if (t > 0) { float vp = x[(size_t)(t - 1) * D + d]; sp += vp * vp; }
    }
    sbuf[tid] = s; __syncthreads();
    for (int off = 128; off > 0; off >>= 1) { if (tid < off) sbuf[tid] += sbuf[tid + off]; __syncthreads(); }
    float rs = rsqrtf(sbuf[0] / (float)D + 1e-6f);
    __syncthreads();
    sbuf[tid] = sp; __syncthreads();
    for (int off = 128; off > 0; off >>= 1) { if (tid < off) sbuf[tid] += sbuf[tid + off]; __syncthreads(); }
    float rsp = rsqrtf(sbuf[0] / (float)D + 1e-6f);

    for (int d = tid; d < D; d += 256) {
        float xnv  = ln1w[d] * x[(size_t)t * D + d] * rs;
        float prev = (t == 0) ? x_prev[d] : ln1w[d] * x[(size_t)(t - 1) * D + d] * rsp;
        float xx   = prev - xnv;
        size_t idx = (size_t)t * D + d;
        xr[idx] = (bf16)(xnv + xx * mr[d]);
        xw[idx] = (bf16)(xnv + xx * mw[d]);
        xk[idx] = (bf16)(xnv + xx * mk[d]);
        xv[idx] = (bf16)(xnv + xx * mv[d]);
        xa[idx] = (bf16)(xnv + xx * ma[d]);
        xg[idx] = (bf16)(xnv + xx * mg[d]);
        if (t == T - 1) xn_last[d] = xnv;
    }
}

// ---------------------------------------------------------------------------
// prep for scan: decay, kk normalize, aa/bb, k & v rework.  one block per row.
// ---------------------------------------------------------------------------
__global__ __launch_bounds__(256)
void prep_scan(float* __restrict__ k, float* __restrict__ v,
               const float* __restrict__ a_pre, const float* __restrict__ vg_pre,
               const float* __restrict__ wv, const float* __restrict__ v_first,
               const float* __restrict__ k_k, const float* __restrict__ k_a,
               const float* __restrict__ w0, const float* __restrict__ a0,
               const float* __restrict__ v0,
               float* __restrict__ decay, float* __restrict__ aa, float* __restrict__ bb,
               int D) {
    const int t = blockIdx.x, tid = threadIdx.x;
    __shared__ float kkbuf[1024];
    __shared__ float hnorm[16];

    for (int d = tid; d < D; d += 256) kkbuf[d] = k[(size_t)t * D + d] * k_k[d];
    __syncthreads();
    if (tid < 16) {
        float s = 0.f;
        for (int j = 0; j < 64; ++j) { float u = kkbuf[tid * 64 + j]; s += u * u; }
        hnorm[tid] = fmaxf(sqrtf(s), 1e-12f);
    }
    __syncthreads();

    for (int d = tid; d < D; d += 256) {
        size_t idx = (size_t)t * D + d;
        int h = d >> 6;
        float kkn = kkbuf[d] / hnorm[h];
        float a_s = 1.f / (1.f + expf(-(a0[d] + a_pre[idx])));
        aa[idx] = -kkn;
        bb[idx] = kkn * a_s;
        float kv = k[idx] * (1.f + (a_s - 1.f) * k_a[d]);
        k[idx] = kv;
        float vg = 1.f / (1.f + expf(-(v0[d] + vg_pre[idx])));
        float vv = v[idx];
        v[idx] = vv + (v_first[idx] - vv) * vg;
        float z  = -(w0[d] + wv[idx]);
        float sp = (z > 20.f) ? z : log1pf(expf(z));
        float wp = -sp - 0.5f;
        decay[idx] = expf(-expf(wp));
    }
}

// ---------------------------------------------------------------------------
// RWKV-7 recurrence. One block per head, 64 threads, state row in VGPRs.
// ---------------------------------------------------------------------------
__global__ __launch_bounds__(64)
void scan_kernel(const float* __restrict__ r, const float* __restrict__ decay,
                 const float* __restrict__ k, const float* __restrict__ v,
                 const float* __restrict__ aa, const float* __restrict__ bb,
                 const float* __restrict__ state_in,
                 float* __restrict__ o, float* __restrict__ state_out,
                 int T, int D) {
    const int h = blockIdx.x, i = threadIdx.x;
    float S[64];
#pragma unroll
    for (int j = 0; j < 64; ++j) S[j] = state_in[(size_t)(h * 64 + i) * 64 + j];

    __shared__ float rv[64], dv[64], kv[64], vv[64], av[64], bv[64];
    for (int t = 0; t < T; ++t) {
        size_t base = (size_t)t * D + h * 64;
        rv[i] = r[base + i]; dv[i] = decay[base + i]; kv[i] = k[base + i];
        vv[i] = v[base + i]; av[i] = aa[base + i];    bv[i] = bb[base + i];
        __syncthreads();
        float sa = 0.f;
#pragma unroll
        for (int j = 0; j < 64; ++j) sa += S[j] * av[j];
        float vi = vv[i], ot = 0.f;
#pragma unroll
        for (int j = 0; j < 64; ++j) {
            S[j] = S[j] * dv[j] + sa * bv[j] + vi * kv[j];
            ot += S[j] * rv[j];
        }
        o[base + i] = ot;
        __syncthreads();
    }
#pragma unroll
    for (int j = 0; j < 64; ++j) state_out[(size_t)(h * 64 + i) * 64 + j] = S[j];
}

// ---------------------------------------------------------------------------
// bonus + gating: o += (sum_h r*k*r_k) * v ; og = bf16(o * g). block per row.
// ---------------------------------------------------------------------------
__global__ __launch_bounds__(256)
void bonus_gate(const float* __restrict__ r, const float* __restrict__ k,
                const float* __restrict__ v, const float* __restrict__ r_k,
                const float* __restrict__ g, float* __restrict__ o,
                bf16* __restrict__ og, int D) {
    const int t = blockIdx.x, tid = threadIdx.x;
    __shared__ float buf[1024];
    __shared__ float hsum[16];
    for (int d = tid; d < D; d += 256) {
        size_t idx = (size_t)t * D + d;
        buf[d] = r[idx] * k[idx] * r_k[d];
    }
    __syncthreads();
    if (tid < 16) {
        float s = 0.f;
        for (int j = 0; j < 64; ++j) s += buf[tid * 64 + j];
        hsum[tid] = s;
    }
    __syncthreads();
    for (int d = tid; d < D; d += 256) {
        size_t idx = (size_t)t * D + d;
        float ov = o[idx] + hsum[d >> 6] * v[idx];
        o[idx] = ov;
        og[idx] = (bf16)(ov * g[idx]);
    }
}

// ---------------------------------------------------------------------------
// x1 = x + o1 ; xm = bf16(rmsnorm(x1, ln2)). block per row.
// ---------------------------------------------------------------------------
__global__ __launch_bounds__(256)
void resid_ln2(const float* __restrict__ x, const float* __restrict__ o1,
               const float* __restrict__ ln2w, float* __restrict__ x1,
               bf16* __restrict__ xm, int D) {
    const int t = blockIdx.x, tid = threadIdx.x;
    __shared__ float sbuf[256];
    float s = 0.f;
    for (int d = tid; d < D; d += 256) {
        size_t idx = (size_t)t * D + d;
        float v = x[idx] + o1[idx];
        x1[idx] = v;
        s += v * v;
    }
    sbuf[tid] = s; __syncthreads();
    for (int off = 128; off > 0; off >>= 1) { if (tid < off) sbuf[tid] += sbuf[tid + off]; __syncthreads(); }
    float rs = rsqrtf(sbuf[0] / (float)D + 1e-6f);
    for (int d = tid; d < D; d += 256) {
        size_t idx = (size_t)t * D + d;
        xm[idx] = (bf16)(ln2w[d] * x1[idx] * rs);
    }
}

__global__ void swiglu_kernel(const float* __restrict__ gt, const float* __restrict__ up,
                              bf16* __restrict__ h, int n) {
    int i = blockIdx.x * blockDim.x + threadIdx.x;
    if (i >= n) return;
    float gv = gt[i];
    float si = gv / (1.f + expf(-gv));
    h[i] = (bf16)(si * up[i]);
}

__global__ void final_add(const float* __restrict__ x1, const float* __restrict__ dn,
                          float* __restrict__ out, int n) {
    int i = blockIdx.x * blockDim.x + threadIdx.x;
    if (i < n) out[i] = x1[i] + dn[i];
}

// ---------------------------------------------------------------------------
extern "C" void kernel_launch(void* const* d_in, const int* in_sizes, int n_in,
                              void* d_out, int out_size, void* d_ws, size_t ws_size,
                              hipStream_t stream) {
    const int T = 2048, D = 1024, H = 16, F = 4096;
    const int WL = 64, AL = 64, VL = 32, GL = 128;

    const float* x       = (const float*)d_in[0];
    const float* x_prev  = (const float*)d_in[1];
    const float* v_first = (const float*)d_in[2];
    const float* state0  = (const float*)d_in[3];
    const float* ln1_w   = (const float*)d_in[4];
    const float* ln2_w   = (const float*)d_in[5];
    const float* x_r     = (const float*)d_in[6];
    const float* x_w     = (const float*)d_in[7];
    const float* x_k     = (const float*)d_in[8];
    const float* x_v     = (const float*)d_in[9];
    const float* x_a     = (const float*)d_in[10];
    const float* x_g     = (const float*)d_in[11];
    const float* w0      = (const float*)d_in[12];
    const float* w1      = (const float*)d_in[13];
    const float* w2      = (const float*)d_in[14];
    const float* a0      = (const float*)d_in[15];
    const float* a1      = (const float*)d_in[16];
    const float* a2      = (const float*)d_in[17];
    const float* v0      = (const float*)d_in[18];
    const float* v1      = (const float*)d_in[19];
    const float* v2      = (const float*)d_in[20];
    const float* g1      = (const float*)d_in[21];
    const float* g2      = (const float*)d_in[22];
    const float* k_k     = (const float*)d_in[23];
    const float* k_a     = (const float*)d_in[24];
    const float* r_k     = (const float*)d_in[25];
    const float* R_      = (const float*)d_in[26];
    const float* K_      = (const float*)d_in[27];
    const float* V_      = (const float*)d_in[28];
    const float* O_      = (const float*)d_in[29];
    const float* gate_w  = (const float*)d_in[30];
    const float* up_w    = (const float*)d_in[31];
    const float* down_w  = (const float*)d_in[32];

    float* out_x     = (float*)d_out;
    float* out_xn    = out_x + (size_t)T * D;
    float* out_state = out_xn + D;

    // ---- bump allocator over d_ws ----
    char*  wsp = (char*)d_ws;
    size_t off = 0;
    auto alloc = [&](size_t bytes) -> void* {
        void* p = wsp + off;
        off += (bytes + 255) & ~(size_t)255;
        return p;
    };

    // bf16 weights
    bf16* Rb    = (bf16*)alloc((size_t)D * D * 2);
    bf16* Kb    = (bf16*)alloc((size_t)D * D * 2);
    bf16* Vb    = (bf16*)alloc((size_t)D * D * 2);
    bf16* Ob    = (bf16*)alloc((size_t)D * D * 2);
    bf16* GateB = (bf16*)alloc((size_t)D * F * 2);
    bf16* UpB   = (bf16*)alloc((size_t)D * F * 2);
    bf16* DownB = (bf16*)alloc((size_t)F * D * 2);
    bf16* w1b   = (bf16*)alloc((size_t)D * WL * 2);
    bf16* w2b   = (bf16*)alloc((size_t)WL * D * 2);
    bf16* a1b   = (bf16*)alloc((size_t)D * AL * 2);
    bf16* a2b   = (bf16*)alloc((size_t)AL * D * 2);
    bf16* v1b   = (bf16*)alloc((size_t)D * VL * 2);
    bf16* v2b   = (bf16*)alloc((size_t)VL * D * 2);
    bf16* g1b   = (bf16*)alloc((size_t)D * GL * 2);
    bf16* g2b   = (bf16*)alloc((size_t)GL * D * 2);

    // bf16 activations
    bf16* xrb = (bf16*)alloc((size_t)T * D * 2);
    bf16* xwb = (bf16*)alloc((size_t)T * D * 2);
    bf16* xkb = (bf16*)alloc((size_t)T * D * 2);
    bf16* xvb = (bf16*)alloc((size_t)T * D * 2);
    bf16* xab = (bf16*)alloc((size_t)T * D * 2);
    bf16* xgb = (bf16*)alloc((size_t)T * D * 2);
    bf16* ogb = (bf16*)alloc((size_t)T * D * 2);

    // f32 activations
    float* r32    = (float*)alloc((size_t)T * D * 4);
    float* k32    = (float*)alloc((size_t)T * D * 4);
    float* v32    = (float*)alloc((size_t)T * D * 4);
    float* wvb    = (float*)alloc((size_t)T * D * 4);  // wv; reused as dn later
    float* a_pre  = (float*)alloc((size_t)T * D * 4);
    float* vg_pre = (float*)alloc((size_t)T * D * 4);
    float* gvec   = (float*)alloc((size_t)T * D * 4);
    float* decay  = (float*)alloc((size_t)T * D * 4);  // reused as o1 later
    float* aav    = (float*)alloc((size_t)T * D * 4);  // reused as x1 later
    float* bbv    = (float*)alloc((size_t)T * D * 4);  // reused as xm_b bytes later
    float* obuf   = (float*)alloc((size_t)T * D * 4);
    float* tbuf   = (float*)alloc((size_t)T * GL * 4); // stage-1 LoRA out (max N=128)
    bf16*  tbb    = (bf16*)alloc((size_t)T * GL * 2);
    float* up32   = (float*)alloc((size_t)T * F * 4);
    float* gt32   = (float*)alloc((size_t)T * F * 4);
    bf16*  hb     = (bf16*)alloc((size_t)T * F * 2);
    (void)ws_size; (void)in_sizes; (void)n_in; (void)out_size;

    auto cvt = [&](const float* src, bf16* dst, size_t n) {
        cvt_f32_bf16<<<dim3((unsigned)((n + 255) / 256)), dim3(256), 0, stream>>>(src, dst, (int)n);
    };
    auto gemm = [&](const bf16* A, const bf16* W, float* C, int M, int N, int K) {
        dim3 g((M + 127) / 128, (N + 127) / 128);
        gemm_bf16_kernel<<<g, dim3(256), 0, stream>>>(A, W, C, M, N, K);
    };

    // ---- weight conversion ----
    cvt(R_, Rb, (size_t)D * D);   cvt(K_, Kb, (size_t)D * D);
    cvt(V_, Vb, (size_t)D * D);   cvt(O_, Ob, (size_t)D * D);
    cvt(gate_w, GateB, (size_t)D * F); cvt(up_w, UpB, (size_t)D * F);
    cvt(down_w, DownB, (size_t)F * D);
    cvt(w1, w1b, (size_t)D * WL); cvt(w2, w2b, (size_t)WL * D);
    cvt(a1, a1b, (size_t)D * AL); cvt(a2, a2b, (size_t)AL * D);
    cvt(v1, v1b, (size_t)D * VL); cvt(v2, v2b, (size_t)VL * D);
    cvt(g1, g1b, (size_t)D * GL); cvt(g2, g2b, (size_t)GL * D);

    // ---- ln1 + token shift ----
    rmsnorm_shift<<<dim3(T), dim3(256), 0, stream>>>(
        x, x_prev, ln1_w, x_r, x_w, x_k, x_v, x_a, x_g,
        xrb, xwb, xkb, xvb, xab, xgb, out_xn, T, D);

    // ---- big projections ----
    gemm(xrb, Rb, r32, T, D, D);
    gemm(xkb, Kb, k32, T, D, D);
    gemm(xvb, Vb, v32, T, D, D);

    // ---- LoRA pipelines ----
    gemm(xwb, w1b, tbuf, T, WL, D);
    act_cvt<<<dim3((T * WL + 255) / 256), dim3(256), 0, stream>>>(tbuf, tbb, T * WL, 1);
    gemm(tbb, w2b, wvb, T, D, WL);

    gemm(xab, a1b, tbuf, T, AL, D);
    act_cvt<<<dim3((T * AL + 255) / 256), dim3(256), 0, stream>>>(tbuf, tbb, T * AL, 0);
    gemm(tbb, a2b, a_pre, T, D, AL);

    gemm(xvb, v1b, tbuf, T, VL, D);
    act_cvt<<<dim3((T * VL + 255) / 256), dim3(256), 0, stream>>>(tbuf, tbb, T * VL, 0);
    gemm(tbb, v2b, vg_pre, T, D, VL);

    gemm(xgb, g1b, tbuf, T, GL, D);
    act_cvt<<<dim3((T * GL + 255) / 256), dim3(256), 0, stream>>>(tbuf, tbb, T * GL, 2);
    gemm(tbb, g2b, gvec, T, D, GL);

    // ---- elementwise prep ----
    prep_scan<<<dim3(T), dim3(256), 0, stream>>>(
        k32, v32, a_pre, vg_pre, wvb, v_first, k_k, k_a, w0, a0, v0,
        decay, aav, bbv, D);

    // ---- recurrence ----
    scan_kernel<<<dim3(H), dim3(64), 0, stream>>>(
        r32, decay, k32, v32, aav, bbv, state0, obuf, out_state, T, D);

    // ---- bonus + gating ----
    bonus_gate<<<dim3(T), dim3(256), 0, stream>>>(
        r32, k32, v32, r_k, gvec, obuf, ogb, D);

    // ---- output projection (reuse decay as o1) ----
    float* o1 = decay;
    gemm(ogb, Ob, o1, T, D, D);

    // ---- residual + ln2 (reuse aav as x1, bbv bytes as xm) ----
    float* x1  = aav;
    bf16*  xmb = (bf16*)bbv;
    resid_ln2<<<dim3(T), dim3(256), 0, stream>>>(x, o1, ln2_w, x1, xmb, D);

    // ---- SwiGLU MLP ----
    gemm(xmb, GateB, gt32, T, F, D);
    gemm(xmb, UpB,   up32, T, F, D);
    swiglu_kernel<<<dim3((T * F + 255) / 256), dim3(256), 0, stream>>>(gt32, up32, hb, T * F);
    float* dn = wvb;
    gemm(hb, DownB, dn, T, D, F);

    // ---- final residual -> d_out ----
    final_add<<<dim3((T * D + 255) / 256), dim3(256), 0, stream>>>(x1, dn, out_x, T * D);
}
